// graphNetwork_14224931684571
// MI455X (gfx1250) — matrix operand order repaired and checked by reading, not compiled, not run on hardware
//
#include <hip/hip_runtime.h>

// ---------------- problem constants (from the reference) ----------------
#define E_EDGES 128
#define DEG     12
#define HH      64
#define PP      (HH * HH)     // 4096 pixels per image

typedef __attribute__((ext_vector_type(16))) _Float16 v16h;
typedef __attribute__((ext_vector_type(8)))  float    v8f;

// =======================================================================
// Kernel 1: build first conv input  X[e, c, p] (f16), c: 0..2 img, 3 mask
// =======================================================================
__global__ __launch_bounds__(256) void build_input_kernel(
    const float* __restrict__ img, const float* __restrict__ masks,
    _Float16* __restrict__ X)
{
    int idx = blockIdx.x * 256 + threadIdx.x;
    if (idx >= E_EDGES * 4 * PP) return;
    int p = idx & (PP - 1);
    int c = (idx >> 12) & 3;
    int e = idx >> 14;
    float v = (c < 3) ? img[c * PP + p] : masks[e * PP + p];
    X[idx] = (_Float16)v;
}

__global__ __launch_bounds__(256) void zero_f32_kernel(float* __restrict__ p, int n)
{
    int i = blockIdx.x * 256 + threadIdx.x;
    if (i < n) p[i] = 0.f;
}

// =======================================================================
// Kernel 2: implicit-GEMM 3x3 SAME conv via v_wmma_f32_16x16x32_f16.
//   in : [E, CIN, 64, 64] f16      w : [Cout, CIN*9] f32 (OIHW flat)
//   out: [E, Cout, 64, 64] f16 (bias added; optional ReLU before store)
//   stats: [2*Cout] f32  (sum, sumsq per channel) if doStats
//
// Block = 128 threads (4 waves, wave32). Wave w owns the 16x16 tile
// (channels co0..co0+15) x (x = 16w..16w+15). Pipelined step loop over
// (8 rows) x (K/32 steps) with a double-buffered im2col tile in LDS:
// one barrier per WMMA. Decode tables remove per-element div/mod; padded
// K entries use ky=100 so the border test also kills them.
// blockIdx = (Cout/16, E, 8 row-groups of 8 rows).
// =======================================================================
template<int CIN>
__global__ __launch_bounds__(128) void conv3x3_wmma_kernel(
    const _Float16* __restrict__ in, const float* __restrict__ w,
    const float* __restrict__ bias, _Float16* __restrict__ out,
    float* __restrict__ stats, int Cout, int reluPre, int doStats)
{
    constexpr int K  = CIN * 9;              // GEMM depth
    constexpr int KP = (K + 31) & ~31;       // padded to WMMA K step
    constexpr int S  = KP / 32;              // K-steps per output row
    constexpr int T  = 8 * S;                // pipelined steps per block

    __shared__ _Float16    Alds[16][KP];     // weight strip (persistent)
    __shared__ _Float16    Blds[2][64][32];  // im2col tiles, [n][k], double buffered
    __shared__ short       tCi[KP];          // k -> input channel
    __shared__ signed char tKy[KP];          // k -> ky-1 (100 => padded/invalid k)
    __shared__ signed char tKx[KP];          // k -> kx-1
    __shared__ float       sSum[16], sSq[16];

    const int tid  = threadIdx.x;
    const int lane = tid & 31;
    const int wv   = tid >> 5;               // wave id 0..3
    const int co0  = blockIdx.x * 16;
    const int e    = blockIdx.y;
    const int y0   = blockIdx.z * 8;

    // ---- one-time block setup: weights, decode tables, stat accumulators
    for (int i = tid; i < 16 * KP; i += 128) {
        int m = i / KP, k = i - m * KP;
        float v = (k < K) ? w[(co0 + m) * K + k] : 0.f;
        Alds[m][k] = (_Float16)v;
    }
    for (int k = tid; k < KP; k += 128) {
        if (k < K) {
            int ci = k / 9, r = k - ci * 9;
            tCi[k] = (short)ci;
            tKy[k] = (signed char)(r / 3 - 1);
            tKx[k] = (signed char)(r % 3 - 1);
        } else {
            tCi[k] = 0; tKy[k] = 100; tKx[k] = 0;   // forces bounds test to fail
        }
    }
    if (tid < 16) { sSum[tid] = 0.f; sSq[tid] = 0.f; }
    __syncthreads();

    const int xw = wv * 16;                  // wave's N (x) offset
    const int m  = lane & 15;                // A row / B column within tile
    const int hi = lane >> 4;                // lane half selector
    const int xb = tid & 63;                 // builder's x column
    const int kb = tid >> 6;                 // builder's starting kk (0/1)
    const size_t inBase = (size_t)e * CIN * PP;

    // per-lane fixed channel set -> hoist bias + output bases + reg stats
    float bv[8], rs[8], rq[8];
    size_t ob[8];
    #pragma unroll
    for (int r = 0; r < 8; ++r) {
        const int co = co0 + r + 8 * hi;
        bv[r] = bias[co];
        ob[r] = ((size_t)(e * Cout + co) * HH) * HH + (xw + m);
        rs[r] = 0.f; rq[r] = 0.f;
    }

    // ---- prologue: build tile for step 0 (y0, k0=0) into buffer 0
    for (int kk = kb; kk < 32; kk += 2) {
        int ys = y0 + tKy[kk], xs = xb + tKx[kk];
        _Float16 v = (_Float16)0.f;
        if (((unsigned)ys < (unsigned)HH) & ((unsigned)xs < (unsigned)HH))
            v = in[inBase + (size_t)tCi[kk] * PP + ys * HH + xs];
        Blds[0][xb][kk] = v;
    }
    __syncthreads();

    v8f acc = {0.f, 0.f, 0.f, 0.f, 0.f, 0.f, 0.f, 0.f};
    for (int t = 0; t < T; ++t) {
        const int cur = t & 1;
        const int s   = t - (t / S) * S;     // k-step within row
        const int k0  = s * 32;
        const int y   = y0 + t / S;

        // ---- read fragments (ISA 7.12.2 16-bit wave32 layouts)
        v16h a, b;
        #pragma unroll
        for (int q = 0; q < 4; ++q) {        // A VGPR 0..3: K = hi*8 + 2q (+1)
            const int k = k0 + hi * 8 + 2 * q;
            a[2 * q]     = Alds[m][k];
            a[2 * q + 1] = Alds[m][k + 1];
        }
        #pragma unroll
        for (int q = 0; q < 4; ++q) {        // A VGPR 4..7: K = 16 + hi*8 + 2q (+1)
            const int k = k0 + 16 + hi * 8 + 2 * q;
            a[8 + 2 * q]     = Alds[m][k];
            a[8 + 2 * q + 1] = Alds[m][k + 1];
        }
        {
            const int koff = hi * 8;         // K = hi*8 + j (j<8); 16 + hi*8 + (j-8) (j>=8)
            #pragma unroll
            for (int j = 0; j < 16; ++j)
                b[j] = Blds[cur][xw + m][koff + j + (j >= 8 ? 8 : 0)];
        }

        // ---- build next tile into the other buffer while this one is consumed
        if (t + 1 < T) {
            const int tn  = t + 1;
            const int k0n = (tn - (tn / S) * S) * 32;
            const int yn  = y0 + tn / S;
            for (int kk = kb; kk < 32; kk += 2) {
                const int k = k0n + kk;
                int ys = yn + tKy[k], xs = xb + tKx[k];
                _Float16 v = (_Float16)0.f;
                if (((unsigned)ys < (unsigned)HH) & ((unsigned)xs < (unsigned)HH))
                    v = in[inBase + (size_t)tCi[k] * PP + ys * HH + xs];
                Blds[cur ^ 1][xb][kk] = v;
            }
        }
        __syncthreads();                     // tile consumed + next tile ready

        acc = __builtin_amdgcn_wmma_f32_16x16x32_f16(
            false, a, false, b, (short)0, acc, false, false);

        // ---- row complete: bias (+ReLU), f16 store, stats into registers
        if (s == S - 1) {
            #pragma unroll
            for (int r = 0; r < 8; ++r) {    // C/D layout: M = r + 8*hi, N = lane&15
                float v = acc[r] + bv[r];
                if (reluPre) v = fmaxf(v, 0.f);
                out[ob[r] + y * HH] = (_Float16)v;
                rs[r] += v; rq[r] += v * v;
            }
            acc = (v8f){0.f, 0.f, 0.f, 0.f, 0.f, 0.f, 0.f, 0.f};
        }
    }

    // ---- flush BN partial stats: regs -> LDS (once) -> global (16 atomics)
    if (doStats) {
        #pragma unroll
        for (int r = 0; r < 8; ++r) {
            atomicAdd(&sSum[r + 8 * hi], rs[r]);
            atomicAdd(&sSq[r + 8 * hi],  rq[r]);
        }
        __syncthreads();
        if (tid < 16) {
            atomicAdd(&stats[co0 + tid],        sSum[tid]);
            atomicAdd(&stats[Cout + co0 + tid], sSq[tid]);
        }
    }
}

// =======================================================================
// Kernel 3: BatchNorm apply (batch stats over E*H*W), optional trailing ReLU
// =======================================================================
__global__ __launch_bounds__(256) void bn_apply_kernel(
    const _Float16* __restrict__ y, const float* __restrict__ stats,
    const float* __restrict__ g, const float* __restrict__ be,
    _Float16* __restrict__ outp, int Cout, int reluPost, int total)
{
    int idx = blockIdx.x * 256 + threadIdx.x;
    if (idx >= total) return;
    int c = (idx >> 12) % Cout;
    const float Ninv = 1.f / (float)(E_EDGES * PP);
    float mean = stats[c] * Ninv;
    float var  = stats[Cout + c] * Ninv - mean * mean;
    float sc   = rsqrtf(var + 1e-5f) * g[c];
    float v    = ((float)y[idx] - mean) * sc + be[c];
    if (reluPost) v = fmaxf(v, 0.f);
    outp[idx] = (_Float16)v;
}

// =======================================================================
// Kernel 4: neighbor segment-max gather + concat -> h[e, 0..63, p]
// (src = repeat(arange(E), DEG), so segment e == nbr[e*DEG .. e*DEG+DEG))
// =======================================================================
__global__ __launch_bounds__(256) void gather_concat_kernel(
    const _Float16* __restrict__ feat, const int* __restrict__ nbr,
    _Float16* __restrict__ h)
{
    int idx = blockIdx.x * 256 + threadIdx.x;
    if (idx >= E_EDGES * 32 * PP) return;
    int p = idx & (PP - 1);
    int c = (idx >> 12) & 31;
    int e = idx >> 17;
    float mx = -3.0e38f;
    #pragma unroll 4
    for (int j = 0; j < DEG; ++j) {
        int ne = nbr[e * DEG + j];
        float v = (float)feat[(ne * 32 + c) * PP + p];
        mx = fmaxf(mx, v);
    }
    h[(e * 64 + c) * PP + p]      = feat[idx];
    h[(e * 64 + 32 + c) * PP + p] = (_Float16)mx;
}

// =======================================================================
// Kernel 5: AdaptiveAvgPool2d(2,2) over 32x32 blocks + FC (512 -> 2)
// One block per edge image.
// =======================================================================
__global__ __launch_bounds__(256) void pool_fc_kernel(
    const _Float16* __restrict__ feat, const float* __restrict__ fw,
    const float* __restrict__ fb, float* __restrict__ outp)
{
    __shared__ float pooled[512];            // [c(128)][qy(2)][qx(2)]
    const int e = blockIdx.x, tid = threadIdx.x;
    for (int i = tid; i < 512; i += 256) {
        int c = i >> 2, qy = (i >> 1) & 1, qx = i & 1;
        const _Float16* base = feat + ((size_t)(e * 128 + c) * HH + qy * 32) * HH + qx * 32;
        float s = 0.f;
        for (int yy = 0; yy < 32; ++yy)
            for (int xx = 0; xx < 32; ++xx)
                s += (float)base[yy * HH + xx];
        pooled[i] = s * (1.f / 1024.f);
    }
    __syncthreads();
    if (tid < 2) {
        float s = fb[tid];
        for (int i = 0; i < 512; ++i) s += pooled[i] * fw[i * 2 + tid];
        outp[e * 2 + tid] = s;
    }
}

// ---------------------------- host side --------------------------------
struct CP { const float *w, *b, *g, *be; };

static void run_conv(int cin, const _Float16* in, const CP& p, _Float16* outp,
                     float* stats, int cout, int relu, int dostats, hipStream_t s)
{
    dim3 grid(cout / 16, E_EDGES, HH / 8), block(128);
    switch (cin) {
    case 4:
        conv3x3_wmma_kernel<4><<<grid, block, 0, s>>>(in, p.w, p.b, outp, stats, cout, relu, dostats);
        break;
    case 32:
        conv3x3_wmma_kernel<32><<<grid, block, 0, s>>>(in, p.w, p.b, outp, stats, cout, relu, dostats);
        break;
    case 64:
        conv3x3_wmma_kernel<64><<<grid, block, 0, s>>>(in, p.w, p.b, outp, stats, cout, relu, dostats);
        break;
    default:
        conv3x3_wmma_kernel<128><<<grid, block, 0, s>>>(in, p.w, p.b, outp, stats, cout, relu, dostats);
        break;
    }
}

extern "C" void kernel_launch(void* const* d_in, const int* in_sizes, int n_in,
                              void* d_out, int out_size, void* d_ws, size_t ws_size,
                              hipStream_t stream)
{
    (void)in_sizes; (void)n_in; (void)out_size; (void)ws_size;

    // -------- inputs (setup_inputs dict order, nested in insertion order) --------
    const float* img   = (const float*)d_in[0];       // [1,3,64,64]
    const float* masks = (const float*)d_in[1];       // [128,64,64]
    const int*   eidx  = (const int*)d_in[2];         // [2, 128*12]
    const int*   nbr   = eidx + E_EDGES * DEG;        // row 1 = neighbor ids

    int pi = 3;
    auto take = [&](CP& c) {
        c.w  = (const float*)d_in[pi++]; c.b  = (const float*)d_in[pi++];
        c.g  = (const float*)d_in[pi++]; c.be = (const float*)d_in[pi++];
    };
    CP backbone; take(backbone);
    CP rgb;      take(rgb);
    CP mpn[2][7];
    for (int t = 0; t < 2; ++t) for (int l = 0; l < 7; ++l) take(mpn[t][l]);
    CP pred[5];
    for (int l = 0; l < 5; ++l) take(pred[l]);
    const float* fcw = (const float*)d_in[pi++];      // [512,2]
    const float* fcb = (const float*)d_in[pi++];      // [2]

    // -------- workspace layout (f16 ping-pong + feat + stats) --------
    const size_t BUF = (size_t)E_EDGES * 128 * PP * sizeof(_Float16);  // 134,217,728 B
    char* ws = (char*)d_ws;
    _Float16* X     = (_Float16*)ws;                      // activations / conv input
    _Float16* Y     = (_Float16*)(ws + BUF);              // conv output (pre-BN)
    _Float16* F     = (_Float16*)(ws + 2 * BUF);          // feat [128,32,4096]
    float*    stats = (float*)   (ws + 2 * BUF + (size_t)E_EDGES * 32 * PP * sizeof(_Float16));

    auto blocks = [](int total) { return (total + 255) / 256; };

    // -------- rgb_net: backbone conv (4->64), rgb conv (64->32), no act/BN --------
    build_input_kernel<<<blocks(E_EDGES * 4 * PP), 256, 0, stream>>>(img, masks, X);
    run_conv(4,  X, backbone, Y, stats, 64, 0, 0, stream);
    run_conv(64, Y, rgb,      F, stats, 32, 0, 0, stream);

    // -------- 2 conv-MPN rounds: gather/max-concat then 7x (Conv->ReLU->BN) --------
    const int mcin[7]  = {64, 64, 64, 64, 64, 32, 32};
    const int mcout[7] = {64, 64, 64, 64, 32, 32, 32};
    for (int t = 0; t < 2; ++t) {
        gather_concat_kernel<<<blocks(E_EDGES * 32 * PP), 256, 0, stream>>>(F, nbr, X);
        for (int l = 0; l < 7; ++l) {
            zero_f32_kernel<<<1, 256, 0, stream>>>(stats, 2 * mcout[l]);
            run_conv(mcin[l], X, mpn[t][l], Y, stats, mcout[l], /*reluPre=*/1, /*stats=*/1, stream);
            _Float16* dst = (l == 6) ? F : X;   // round result -> feat buffer
            int total = E_EDGES * mcout[l] * PP;
            bn_apply_kernel<<<blocks(total), 256, 0, stream>>>(
                Y, stats, mpn[t][l].g, mpn[t][l].be, dst, mcout[l], /*reluPost=*/0, total);
        }
    }

    // -------- edge_pred: 5x (Conv->BN->ReLU) --------
    const int pcin[5]  = {32, 32, 64, 64, 128};
    const int pcout[5] = {32, 64, 64, 128, 128};
    const _Float16* src = F;
    for (int l = 0; l < 5; ++l) {
        zero_f32_kernel<<<1, 256, 0, stream>>>(stats, 2 * pcout[l]);
        run_conv(pcin[l], src, pred[l], Y, stats, pcout[l], /*reluPre=*/0, /*stats=*/1, stream);
        int total = E_EDGES * pcout[l] * PP;
        bn_apply_kernel<<<blocks(total), 256, 0, stream>>>(
            Y, stats, pred[l].g, pred[l].be, X, pcout[l], /*reluPost=*/1, total);
        src = X;
    }

    // -------- 2x2 avg pool + FC --------
    pool_fc_kernel<<<E_EDGES, 256, 0, stream>>>(X, fcw, fcb, (float*)d_out);
}
